// CrossAttention_56599079026812
// MI455X (gfx1250) — compile-verified, matrix operand order checked by reading, not run on hardware
//
#include <hip/hip_runtime.h>
#include <hip/hip_bf16.h>

typedef _Float16 v16h __attribute__((ext_vector_type(16)));
typedef _Float16 v8h  __attribute__((ext_vector_type(8)));
typedef float    v8f  __attribute__((ext_vector_type(8)));

#define D_MODEL  512
#define NHEAD    8
#define HEAD_DIM 64
#define BATCH    2
#define SQL      2048
#define SKV      4096

// ---------------------------------------------------------------------------
// WMMA fragment loader for a row-major matrix (leading dim `ld`, elements):
//   lane L: r = row0 + (L&15), kb = (L>>4)*8
//   frag[i]   = M[r][kb+i]    frag[8+i] = M[r][16+kb+i]   (two b128 loads)
// Works for global or LDS-derived pointers (addrspace inference -> ds_load).
// ---------------------------------------------------------------------------
__device__ __forceinline__ v16h load_frag(const _Float16* p, int ld, int row0) {
  const int lane = threadIdx.x & 31;
  const int r    = row0 + (lane & 15);
  const int kb   = (lane >> 4) << 3;
  const _Float16* base = p + (size_t)r * (size_t)ld + kb;
  v8h lo = *(const v8h*)(base);
  v8h hi = *(const v8h*)(base + 16);
  return __builtin_shufflevector(lo, hi, 0,1,2,3,4,5,6,7,8,9,10,11,12,13,14,15);
}

__device__ __forceinline__ v8f wmma_f16(v16h a, v16h b, v8f c) {
  return __builtin_amdgcn_wmma_f32_16x16x32_f16(false, a, false, b,
                                                (short)0, c, false, false);
}

// ---------------------------------------------------------------------------
// Cross-lane reductions within each 16-lane half (rows of the C layout).
// v_permlane16_b32 does an arbitrary permute inside each row of 16 lanes;
// nibble k of {hi,lo} selects the source lane for lane k  ->  sel[k] = k ^ m.
// ---------------------------------------------------------------------------
#if __has_builtin(__builtin_amdgcn_permlane16)
#define PL16(x, lo, hi)                                                        \
  __int_as_float(__builtin_amdgcn_permlane16(                                  \
      __float_as_int(x), __float_as_int(x), (lo), (hi), false, false))
__device__ __forceinline__ float red_max16(float x) {
  x = fmaxf(x, PL16(x, 0x67452301u, 0xEFCDAB89u));  // xor 1
  x = fmaxf(x, PL16(x, 0x54761032u, 0xDCFE98BAu));  // xor 2
  x = fmaxf(x, PL16(x, 0x32107654u, 0xBA98FEDCu));  // xor 4
  x = fmaxf(x, PL16(x, 0xFEDCBA98u, 0x76543210u));  // xor 8
  return x;
}
__device__ __forceinline__ float red_sum16(float x) {
  x += PL16(x, 0x67452301u, 0xEFCDAB89u);
  x += PL16(x, 0x54761032u, 0xDCFE98BAu);
  x += PL16(x, 0x32107654u, 0xBA98FEDCu);
  x += PL16(x, 0xFEDCBA98u, 0x76543210u);
  return x;
}
#else
__device__ __forceinline__ float red_max16(float x) {
  x = fmaxf(x, __shfl_xor(x, 1, 32));
  x = fmaxf(x, __shfl_xor(x, 2, 32));
  x = fmaxf(x, __shfl_xor(x, 4, 32));
  x = fmaxf(x, __shfl_xor(x, 8, 32));
  return x;
}
__device__ __forceinline__ float red_sum16(float x) {
  x += __shfl_xor(x, 1, 32);
  x += __shfl_xor(x, 2, 32);
  x += __shfl_xor(x, 4, 32);
  x += __shfl_xor(x, 8, 32);
  return x;
}
#endif

// ---------------------------------------------------------------------------
// Async global -> LDS copy, 16 bytes per lane (CDNA5 GLOBAL_LOAD_ASYNC_TO_LDS).
// lds_off = workgroup-relative LDS byte offset (low 32 bits of flat address).
// ---------------------------------------------------------------------------
__device__ __forceinline__ unsigned lds_byte_off(const void* p) {
  return (unsigned)(uintptr_t)p;
}
__device__ __forceinline__ void async_copy_b128(unsigned lds_off,
                                                const void* gptr) {
  asm volatile("global_load_async_to_lds_b128 %0, %1, off"
               :: "v"(lds_off), "v"(gptr) : "memory");
}

// ---------------------------------------------------------------------------
// fp32 -> f16 conversion (grid-stride)
// ---------------------------------------------------------------------------
__global__ void ca_cvt_f32_f16(const float* __restrict__ in,
                               _Float16* __restrict__ out, int n) {
  int i = blockIdx.x * blockDim.x + threadIdx.x;
  int stride = gridDim.x * blockDim.x;
  for (; i < n; i += stride) out[i] = (_Float16)in[i];
}

// ---------------------------------------------------------------------------
// C[M,N] = A[M,K] @ W[N,K]^T + bias ; wave tile = 16x64 (A-frag reused x4)
// ---------------------------------------------------------------------------
template <typename OutT>
__global__ __launch_bounds__(128) void ca_gemm_wmma(
    const _Float16* __restrict__ A, const _Float16* __restrict__ W,
    const float* __restrict__ bias, OutT* __restrict__ C,
    int M, int N, int K) {
  const int wid  = threadIdx.x >> 5;
  const int lane = threadIdx.x & 31;
  const int wg   = blockIdx.x * 4 + wid;
  const int tiles_n = N >> 6;
  const int row0 = (wg / tiles_n) * 16;
  const int col0 = (wg % tiles_n) * 64;

  v8f acc0 = {}, acc1 = {}, acc2 = {}, acc3 = {};
  for (int k0 = 0; k0 < K; k0 += 32) {
    const v16h a = load_frag(A + k0, K, row0);
    acc0 = wmma_f16(a, load_frag(W + k0, K, col0 +  0), acc0);
    acc1 = wmma_f16(a, load_frag(W + k0, K, col0 + 16), acc1);
    acc2 = wmma_f16(a, load_frag(W + k0, K, col0 + 32), acc2);
    acc3 = wmma_f16(a, load_frag(W + k0, K, col0 + 48), acc3);
  }
  const int half = lane >> 4;
  const int nl   = lane & 15;
  v8f* accs[4] = {&acc0, &acc1, &acc2, &acc3};
#pragma unroll
  for (int j = 0; j < 4; ++j) {
    const int n = col0 + j * 16 + nl;
    const float bv = bias[n];
#pragma unroll
    for (int v = 0; v < 8; ++v) {
      const int m = row0 + v + 8 * half;   // C layout: vgpr v -> row v + 8*half
      C[(size_t)m * N + n] = (OutT)((*accs[j])[v] + bv);
    }
  }
}

// qp[B*SQ, 512] -> qph[B,H,SQ,64]
__global__ void ca_split_q(const _Float16* __restrict__ qp,
                           _Float16* __restrict__ qph, int n) {
  int i = blockIdx.x * blockDim.x + threadIdx.x;
  int stride = gridDim.x * blockDim.x;
  for (; i < n; i += stride) {
    const int d  = i & 63;
    const int sq = (i >> 6) & (SQL - 1);
    const int h  = (i >> 17) & (NHEAD - 1);
    const int b  = i >> 20;
    qph[i] = qp[(size_t)(b * SQL + sq) * D_MODEL + h * HEAD_DIM + d];
  }
}

// kvp[B*SKV, 1024] -> k[B,H,SKV,64] and v^T[B,H,64,SKV]
__global__ void ca_split_kv(const _Float16* __restrict__ kvp,
                            _Float16* __restrict__ kh,
                            _Float16* __restrict__ vt, int n) {
  int i = blockIdx.x * blockDim.x + threadIdx.x;
  int stride = gridDim.x * blockDim.x;
  for (; i < n; i += stride) {
    const int ncol = i & 1023;
    const int row  = i >> 10;
    const int skv  = row & (SKV - 1);
    const int b    = row >> 12;
    const int h    = ncol >> 7;
    const int c    = ncol & 127;
    const int bh   = b * NHEAD + h;
    const _Float16 val = kvp[i];
    if (c < HEAD_DIM)
      kh[((size_t)bh * SKV + skv) * HEAD_DIM + c] = val;
    else
      vt[((size_t)bh * HEAD_DIM + (c - HEAD_DIM)) * SKV + skv] = val;
  }
}

// ---------------------------------------------------------------------------
// Flash attention.  4 waves/block share one (b,h); per 32-row KV step the
// block stages K(32x64) and V^T(64x32) tiles into LDS with
// global_load_async_to_lds_b128, then each wave runs 4 WMMA for S(16x32),
// an online softmax (permlane16 row reductions), a C->A layout transpose of
// P through per-wave LDS, and 4 WMMA for O += P@V.
// ---------------------------------------------------------------------------
__global__ __launch_bounds__(128) void ca_flash_attn(
    const _Float16* __restrict__ qph, const _Float16* __restrict__ kh,
    const _Float16* __restrict__ vt, _Float16* __restrict__ vals) {
  __shared__ __align__(16) _Float16 ktile[32 * HEAD_DIM];   // [kv 32][d 64]
  __shared__ __align__(16) _Float16 vtile[HEAD_DIM * 32];   // [d 64][kv 32]
  __shared__ __align__(16) _Float16 pbuf[4][16][32];

  const int tid  = threadIdx.x;
  const int wid  = tid >> 5;
  const int lane = tid & 31;
  const int wg   = blockIdx.x * 4 + wid;
  const int qt   = wg & (SQL / 16 - 1);   // 128 q-tiles per (b,h); block-contig
  const int bh   = wg >> 7;               // identical across the 4 waves
  const int q0   = qt * 16;
  const int half = lane >> 4;
  const int nl   = lane & 15;
  const int kb   = half << 3;

  const _Float16* Q  = qph + ((size_t)bh * SQL + q0) * HEAD_DIM;
  const _Float16* Kp = kh + (size_t)bh * SKV * HEAD_DIM;
  const _Float16* Vt = vt + (size_t)bh * HEAD_DIM * SKV;

  // Per-thread async-copy slices: 256 chunks of 16B per tile, 2 per thread.
  const unsigned lkA = lds_byte_off(ktile + tid * 8);
  const unsigned lkB = lds_byte_off(ktile + (tid + 128) * 8);
  const unsigned lvA = lds_byte_off(vtile + tid * 8);
  const unsigned lvB = lds_byte_off(vtile + (tid + 128) * 8);
  const _Float16* kgA = Kp + tid * 8;                       // + kv0*64 / iter
  const _Float16* kgB = Kp + (tid + 128) * 8;
  const _Float16* vgA = Vt + (size_t)(tid >> 2) * SKV + (tid & 3) * 8;  // +kv0
  const _Float16* vgB = vgA + (size_t)32 * SKV;

  const v16h qa0 = load_frag(Q, HEAD_DIM, 0);        // d = 0..31
  const v16h qa1 = load_frag(Q + 32, HEAD_DIM, 0);   // d = 32..63

  v8f o0 = {}, o1 = {}, o2 = {}, o3 = {};
  float mrow[8], lrow[8];
#pragma unroll
  for (int v = 0; v < 8; ++v) { mrow[v] = -1e30f; lrow[v] = 0.f; }

  const float cscale = 0.125f * 1.44269504088896f;   // 1/sqrt(64) * log2(e)

  for (int kv0 = 0; kv0 < SKV; kv0 += 32) {
    __builtin_prefetch(kgA + (size_t)(kv0 + 32) * HEAD_DIM, 0, 0);
    async_copy_b128(lkA, kgA + (size_t)kv0 * HEAD_DIM);
    async_copy_b128(lkB, kgB + (size_t)kv0 * HEAD_DIM);
    async_copy_b128(lvA, vgA + kv0);
    async_copy_b128(lvB, vgB + kv0);
    asm volatile("s_wait_asynccnt 0" ::: "memory");
    __syncthreads();

    // S = Q @ K^T for 32 KV columns (B-frags straight from the LDS K tile)
    v8f s0 = {}, s1 = {};
    s0 = wmma_f16(qa0, load_frag(ktile,      HEAD_DIM, 0),  s0);
    s0 = wmma_f16(qa1, load_frag(ktile + 32, HEAD_DIM, 0),  s0);
    s1 = wmma_f16(qa0, load_frag(ktile,      HEAD_DIM, 16), s1);
    s1 = wmma_f16(qa1, load_frag(ktile + 32, HEAD_DIM, 16), s1);

#pragma unroll
    for (int v = 0; v < 8; ++v) {
      const float t0 = s0[v] * cscale;
      const float t1 = s1[v] * cscale;
      const float cur  = red_max16(fmaxf(t0, t1));  // row v+8*half, this half
      const float mnew = fmaxf(mrow[v], cur);
      const float corr = exp2f(mrow[v] - mnew);
      mrow[v] = mnew;
      const float p0 = exp2f(t0 - mnew);
      const float p1 = exp2f(t1 - mnew);
      lrow[v] = lrow[v] * corr + p0 + p1;           // lane-partial row sum
      o0[v] = o0[v] * corr; o1[v] = o1[v] * corr;
      o2[v] = o2[v] * corr; o3[v] = o3[v] * corr;
      pbuf[wid][v + 8 * half][nl]      = (_Float16)p0;   // C-layout -> LDS
      pbuf[wid][v + 8 * half][16 + nl] = (_Float16)p1;
    }
    asm volatile("s_wait_dscnt 0" ::: "memory");     // cross-lane LDS RAW
    const v8h plo = *(const v8h*)&pbuf[wid][nl][kb]; // A-layout reload
    const v8h phi = *(const v8h*)&pbuf[wid][nl][16 + kb];
    const v16h pa = __builtin_shufflevector(plo, phi,
        0,1,2,3,4,5,6,7,8,9,10,11,12,13,14,15);

    // O += P @ V (B-frags from the LDS V^T tile, ld = 32)
    o0 = wmma_f16(pa, load_frag(vtile, 32,  0), o0);
    o1 = wmma_f16(pa, load_frag(vtile, 32, 16), o1);
    o2 = wmma_f16(pa, load_frag(vtile, 32, 32), o2);
    o3 = wmma_f16(pa, load_frag(vtile, 32, 48), o3);
    __syncthreads();   // all waves done with tiles before next async overwrite
  }

#pragma unroll
  for (int v = 0; v < 8; ++v) lrow[v] = 1.0f / red_sum16(lrow[v]);

  const int b = bh >> 3;
  const int h = bh & 7;
  _Float16* outp = vals + (size_t)b * SQL * D_MODEL + (size_t)h * HEAD_DIM;
#pragma unroll
  for (int v = 0; v < 8; ++v) {
    const size_t ro = (size_t)(q0 + v + 8 * half) * D_MODEL;
    outp[ro +  0 + nl] = (_Float16)(o0[v] * lrow[v]);
    outp[ro + 16 + nl] = (_Float16)(o1[v] * lrow[v]);
    outp[ro + 32 + nl] = (_Float16)(o2[v] * lrow[v]);
    outp[ro + 48 + nl] = (_Float16)(o3[v] * lrow[v]);
  }
}

// ---------------------------------------------------------------------------
extern "C" void kernel_launch(void* const* d_in, const int* in_sizes, int n_in,
                              void* d_out, int out_size, void* d_ws, size_t ws_size,
                              hipStream_t stream) {
  (void)in_sizes; (void)n_in; (void)out_size; (void)ws_size;
  const float* q   = (const float*)d_in[0];
  const float* kv  = (const float*)d_in[1];
  const float* Wq  = (const float*)d_in[2];
  const float* bq  = (const float*)d_in[3];
  const float* Wkv = (const float*)d_in[4];
  const float* bkv = (const float*)d_in[5];
  const float* Wo  = (const float*)d_in[6];
  const float* bo  = (const float*)d_in[7];
  float* out = (float*)d_out;

  _Float16* w = (_Float16*)d_ws;                 // offsets in f16 elements
  _Float16* Wq_h   = w;                          //   262144
  _Float16* Wkv_h  = w + 262144;                 //   524288
  _Float16* Wo_h   = w + 786432;                 //   262144
  _Float16* q_h    = w + 1048576;                //  2097152
  _Float16* kv_h   = w + 3145728;                //  4194304
  _Float16* qp_t   = w + 7340032;                //  2097152
  _Float16* kvp_t  = w + 9437184;                //  8388608
  _Float16* qph    = w + 17825792;               //  2097152
  _Float16* k_h    = w + 19922944;               //  4194304
  _Float16* v_t    = w + 24117248;               //  4194304
  _Float16* vals_h = w + 28311552;               //  2097152  (~58 MB total)

  ca_cvt_f32_f16<<<1024, 256, 0, stream>>>(q,   q_h,   BATCH * SQL * D_MODEL);
  ca_cvt_f32_f16<<<1024, 256, 0, stream>>>(kv,  kv_h,  BATCH * SKV * D_MODEL);
  ca_cvt_f32_f16<<<256,  256, 0, stream>>>(Wq,  Wq_h,  D_MODEL * D_MODEL);
  ca_cvt_f32_f16<<<512,  256, 0, stream>>>(Wkv, Wkv_h, 2 * D_MODEL * D_MODEL);
  ca_cvt_f32_f16<<<256,  256, 0, stream>>>(Wo,  Wo_h,  D_MODEL * D_MODEL);

  ca_gemm_wmma<_Float16><<<512,  128, 0, stream>>>(q_h,  Wq_h,  bq,  qp_t,
                                                   BATCH * SQL, D_MODEL, D_MODEL);
  ca_gemm_wmma<_Float16><<<2048, 128, 0, stream>>>(kv_h, Wkv_h, bkv, kvp_t,
                                                   BATCH * SKV, 2 * D_MODEL, D_MODEL);

  ca_split_q<<<1024, 256, 0, stream>>>(qp_t, qph, BATCH * SQL * D_MODEL);
  ca_split_kv<<<1024, 256, 0, stream>>>(kvp_t, k_h, v_t, BATCH * SKV * 2 * D_MODEL);

  ca_flash_attn<<<512, 128, 0, stream>>>(qph, k_h, v_t, vals_h);

  ca_gemm_wmma<float><<<512, 128, 0, stream>>>(vals_h, Wo_h, bo, out,
                                               BATCH * SQL, D_MODEL, D_MODEL);
}
